// Transformer_sd_82927228551542
// MI455X (gfx1250) — matrix operand order repaired and checked by reading, not compiled, not run on hardware
//
#include <hip/hip_runtime.h>
#include <math.h>

// ---------------- CDNA5 WMMA types ----------------
typedef __attribute__((ext_vector_type(16))) __bf16 v16bf;
typedef __attribute__((ext_vector_type(8)))  float  v8f;

union Frag16 { v16bf v; uint4 q[2]; };

__device__ __forceinline__ unsigned int f2bf(float f) {
    unsigned int u = __float_as_uint(f);
    u += 0x7FFFu + ((u >> 16) & 1u);      // round-to-nearest-even
    return u >> 16;
}
__device__ __forceinline__ unsigned int pk2(float a, float b) {
    return f2bf(a) | (f2bf(b) << 16);
}

// ---- CDNA5 async global->LDS (ASYNCcnt-tracked, no VGPR round-trip) ----
__device__ __forceinline__ unsigned lds_off32(const void* p) {
    return (unsigned)(unsigned long long)p;   // LDS generic addr: low 32 bits = offset
}
__device__ __forceinline__ void async_ld_b128(const void* g, unsigned lds) {
    asm volatile("global_load_async_to_lds_b128 %0, %1, off"
                 :: "v"(lds), "v"(g) : "memory");
}
__device__ __forceinline__ void async_ld_b128_o16(const void* g, unsigned lds) {
    // offset:16 applies to BOTH global and LDS address (ISA 08_async_tensor 4.4)
    asm volatile("global_load_async_to_lds_b128 %0, %1, off offset:16"
                 :: "v"(lds), "v"(g) : "memory");
}
__device__ __forceinline__ void wait_async(int n) {
    if (n == 0) asm volatile("s_wait_asynccnt 0x0" ::: "memory");
    else        asm volatile("s_wait_asynccnt 0x3" ::: "memory");
}

#define B_     16
#define NSEQ   273
#define DIM_   768
#define HEADS_ 12
#define DH_    64
#define MLP_   3072
#define J_     17
#define NROWS  (B_ * NSEQ)                 // 4368
#define DEPTH_ 12

typedef unsigned short ushort_t;

// ---------------- elementwise: x[:, J:, :] += pos ----------------
__global__ void __launch_bounds__(256) k_addpos(float* __restrict__ x,
                                                const float* __restrict__ pos) {
    int idx = blockIdx.x * 256 + threadIdx.x;       // 16*256*768 total, exact
    int c = idx % DIM_;
    int t = (idx / DIM_) % (NSEQ - J_);
    int b = idx / (DIM_ * (NSEQ - J_));
    x[((size_t)(b * NSEQ + J_ + t)) * DIM_ + c] += pos[(size_t)t * DIM_ + c];
}

// ---------------- layernorm: one row (768) per block; bf16 output ----------------
__global__ void __launch_bounds__(256) k_layernorm(const float* __restrict__ x,
                                                   const float* __restrict__ w,
                                                   const float* __restrict__ b,
                                                   ushort_t* __restrict__ h) {
    __shared__ float red[256];
    int tid = threadIdx.x;
    const float* xr = x + (size_t)blockIdx.x * DIM_;
    float s = 0.f;
    for (int c = tid; c < DIM_; c += 256) s += xr[c];
    red[tid] = s; __syncthreads();
    for (int off = 128; off > 0; off >>= 1) {
        if (tid < off) red[tid] += red[tid + off];
        __syncthreads();
    }
    float mu = red[0] * (1.f / DIM_);
    __syncthreads();
    float s2 = 0.f;
    for (int c = tid; c < DIM_; c += 256) { float d = xr[c] - mu; s2 += d * d; }
    red[tid] = s2; __syncthreads();
    for (int off = 128; off > 0; off >>= 1) {
        if (tid < off) red[tid] += red[tid + off];
        __syncthreads();
    }
    float rstd = rsqrtf(red[0] * (1.f / DIM_) + 1e-5f);
    ushort_t* hr = h + (size_t)blockIdx.x * DIM_;
    for (int c = tid; c < DIM_; c += 256)
        hr[c] = (ushort_t)f2bf((xr[c] - mu) * rstd * w[c] + b[c]);
}

// ---------------- weight convert + transpose: W[KxN] f32 -> Bt[NxK] bf16 ----------
__global__ void __launch_bounds__(256) k_wtrans(const float* __restrict__ W,
                                                ushort_t* __restrict__ Bt,
                                                int Kd, int Nd) {
    __shared__ ushort_t T[32][34];
    int k0 = blockIdx.y * 32, n0 = blockIdx.x * 32;
    int t = threadIdx.x;
    int r = t >> 3, c4 = (t & 7) * 4;
    float4 f = *(const float4*)(W + (size_t)(k0 + r) * Nd + n0 + c4);
    T[r][c4 + 0] = (ushort_t)f2bf(f.x);
    T[r][c4 + 1] = (ushort_t)f2bf(f.y);
    T[r][c4 + 2] = (ushort_t)f2bf(f.z);
    T[r][c4 + 3] = (ushort_t)f2bf(f.w);
    __syncthreads();
    int nl = t >> 3, k4 = (t & 7) * 4;
    unsigned int lo = (unsigned int)T[k4 + 0][nl] | ((unsigned int)T[k4 + 1][nl] << 16);
    unsigned int hi = (unsigned int)T[k4 + 2][nl] | ((unsigned int)T[k4 + 3][nl] << 16);
    *(uint2*)&Bt[(size_t)(n0 + nl) * Kd + k0 + k4] = make_uint2(lo, hi);
}

// ---------------- fused GEMM: C = act(A@B + bias) [+ Res] ----------------
// A [M x K] bf16 row-major, Bt [N x K] bf16 row-major (pre-transposed weights).
// Block tile 128x64, BK=32; 8 waves -> 2x2 16x16 WMMA tiles each.
// Double-buffered LDS filled by global_load_async_to_lds_b128 (ASYNCcnt).
// OOB A rows are CLAMPED (their C rows are never stored), so no zero-fill.
__global__ void __launch_bounds__(256) k_gemm(const ushort_t* __restrict__ A,
                                              const ushort_t* __restrict__ Bt,
                                              const float* __restrict__ bias,
                                              const float* __restrict__ Res,
                                              float* __restrict__ Cf,
                                              ushort_t* __restrict__ Cb,
                                              int M, int Kd, int Nd, int do_gelu) {
    __shared__ __attribute__((aligned(16))) unsigned int AsU[2][128 * 20];
    __shared__ __attribute__((aligned(16))) unsigned int BsU[2][64 * 20];
    int tid = threadIdx.x, lane = tid & 31, wave = tid >> 5;
    int waveM = wave >> 1, waveN = wave & 1;
    int bm = blockIdx.y * 128, bn = blockIdx.x * 64;

    int arow = tid >> 1, ahalf = tid & 1;          // A stage: one row-half each
    int agrow = bm + arow;
    int aclamp = agrow < M ? agrow : (M - 1);      // clamp: OOB rows never stored
    const ushort_t* aptr = A + (size_t)aclamp * Kd + ahalf * 16;
    int bcol = tid >> 2, bq = tid & 3;             // B stage: col bcol, K bq*8..+8
    const ushort_t* bptr = Bt + (size_t)(bn + bcol) * Kd + bq * 8;

    unsigned lA[2], lB[2];
    #pragma unroll
    for (int i = 0; i < 2; ++i) {
        lA[i] = lds_off32(&AsU[i][arow * 20 + ahalf * 8]);
        lB[i] = lds_off32(&BsU[i][bcol * 20 + bq * 4]);
    }

    int nt = Kd >> 5;                               // K tiles of 32
    // prologue: tile 0 -> buffer 0
    async_ld_b128(aptr, lA[0]);
    async_ld_b128_o16(aptr, lA[0]);
    async_ld_b128(bptr, lB[0]);

    v8f acc[2][2] = {};
    for (int t = 0; t < nt; ++t) {
        int buf = t & 1;
        if (t + 1 < nt) {                           // prefetch next tile
            const ushort_t* ga = aptr + (t + 1) * 32;
            const ushort_t* gb = bptr + (t + 1) * 32;
            async_ld_b128(ga, lA[buf ^ 1]);
            async_ld_b128_o16(ga, lA[buf ^ 1]);
            async_ld_b128(gb, lB[buf ^ 1]);
            wait_async(3);                          // tile t's 3 ops landed
        } else {
            wait_async(0);
        }
        __syncthreads();

        Frag16 af[2], bfv[2];
        int ku = (lane >> 4) * 4;     // A: uint offset of this lane-half's K run
        int kb = (lane >> 4) * 4;     // B: uint offset of this lane-half's K run
        #pragma unroll
        for (int tm = 0; tm < 2; ++tm) {
            int row = waveM * 32 + tm * 16 + (lane & 15);
            af[tm].q[0] = *(const uint4*)&AsU[buf][row * 20 + ku];
            af[tm].q[1] = *(const uint4*)&AsU[buf][row * 20 + 8 + ku];
        }
        #pragma unroll
        for (int tn = 0; tn < 2; ++tn) {
            int col = waveN * 32 + tn * 16 + (lane & 15);
            bfv[tn].q[0] = *(const uint4*)&BsU[buf][col * 20 + kb];
            bfv[tn].q[1] = *(const uint4*)&BsU[buf][col * 20 + kb + 8];
        }
        #pragma unroll
        for (int tm = 0; tm < 2; ++tm)
            #pragma unroll
            for (int tn = 0; tn < 2; ++tn)
                acc[tm][tn] = __builtin_amdgcn_wmma_f32_16x16x32_bf16(
                    false, af[tm].v, false, bfv[tn].v, (short)0, acc[tm][tn], false, false);
        __syncthreads();                            // readers done before refill
    }
    #pragma unroll
    for (int tm = 0; tm < 2; ++tm) {
        #pragma unroll
        for (int tn = 0; tn < 2; ++tn) {
            int colb = bn + waveN * 32 + tn * 16 + (lane & 15);
            #pragma unroll
            for (int r = 0; r < 8; ++r) {
                int grow = bm + waveM * 32 + tm * 16 + r + ((lane >> 4) << 3);
                if (grow < M) {
                    float v = acc[tm][tn][r];
                    if (bias)    v += bias[colb];
                    if (do_gelu) v = 0.5f * v * (1.f + erff(v * 0.70710678118654752f));
                    if (Res)     v += Res[(size_t)grow * Nd + colb];
                    if (Cb) Cb[(size_t)grow * Nd + colb] = (ushort_t)f2bf(v);
                    else    Cf[(size_t)grow * Nd + colb] = v;
                }
            }
        }
    }
}

// ---------------- attention scores: q @ k^T * scale -> attn slice of d_out ----------
// qkv is bf16; B operand (k^T) fragment == k's natural row-major layout.
// OOB rows clamped (their score rows/cols are never stored).
__global__ void __launch_bounds__(256) k_scores(const ushort_t* __restrict__ qkv,
                                                float* __restrict__ attn) {
    __shared__ __attribute__((aligned(16))) unsigned int Qs[64 * 36]; // 64 rows x 32u
    __shared__ __attribute__((aligned(16))) unsigned int Ks[64 * 36];
    int tid = threadIdx.x, lane = tid & 31, wave = tid >> 5;
    int waveM = wave >> 1, waveN = wave & 1;
    int tI = blockIdx.y * 64, tJ = blockIdx.x * 64;
    int bh = blockIdx.z, b = bh / HEADS_, hd = bh % HEADS_;
    const ushort_t* qb = qkv + (size_t)b * NSEQ * (3 * DIM_) + hd * DH_;
    const ushort_t* kb = qb + DIM_;

    {   // stage: each thread one (row, 16-elem quarter) of both Q and K tiles
        int r = tid >> 2, q = tid & 3;
        int gi = tI + r; if (gi >= NSEQ) gi = NSEQ - 1;
        int gj = tJ + r; if (gj >= NSEQ) gj = NSEQ - 1;
        const uint4* pq = (const uint4*)(qb + (size_t)gi * (3 * DIM_) + q * 16);
        const uint4* pk = (const uint4*)(kb + (size_t)gj * (3 * DIM_) + q * 16);
        uint4* dq = (uint4*)&Qs[r * 36 + q * 8];
        dq[0] = pq[0]; dq[1] = pq[1];
        uint4* dk = (uint4*)&Ks[r * 36 + q * 8];
        dk[0] = pk[0]; dk[1] = pk[1];
    }
    __syncthreads();

    v8f acc[2] = {};
    int ku = (lane >> 4) * 4, kbo = (lane >> 4) * 4;
    #pragma unroll
    for (int k0 = 0; k0 < 32; k0 += 16) {           // k0 in uints (=32 bf16/step)
        Frag16 af, bfv;
        int row = waveM * 16 + (lane & 15);
        af.q[0] = *(const uint4*)&Qs[row * 36 + k0 + ku];
        af.q[1] = *(const uint4*)&Qs[row * 36 + k0 + 8 + ku];
        #pragma unroll
        for (int tn = 0; tn < 2; ++tn) {
            int col = waveN * 32 + tn * 16 + (lane & 15);
            bfv.q[0] = *(const uint4*)&Ks[col * 36 + k0 + kbo];
            bfv.q[1] = *(const uint4*)&Ks[col * 36 + k0 + kbo + 8];
            acc[tn] = __builtin_amdgcn_wmma_f32_16x16x32_bf16(
                false, af.v, false, bfv.v, (short)0, acc[tn], false, false);
        }
    }
    float* out = attn + (size_t)bh * NSEQ * NSEQ;
    #pragma unroll
    for (int tn = 0; tn < 2; ++tn) {
        int gj = tJ + waveN * 32 + tn * 16 + (lane & 15);
        #pragma unroll
        for (int r = 0; r < 8; ++r) {
            int gi = tI + waveM * 16 + r + ((lane >> 4) << 3);
            if (gi < NSEQ && gj < NSEQ)
                out[(size_t)gi * NSEQ + gj] = acc[tn][r] * 0.125f;   // 64^-0.5
        }
    }
}

// ---------------- softmax in place, one row of 273 per block ----------------
__global__ void __launch_bounds__(256) k_softmax(float* __restrict__ attn) {
    __shared__ float red[256];
    size_t base = (size_t)blockIdx.x * NSEQ;
    int tid = threadIdx.x;
    float v0 = (tid       < NSEQ) ? attn[base + tid]       : -1e30f;
    float v1 = (tid + 256 < NSEQ) ? attn[base + tid + 256] : -1e30f;
    red[tid] = fmaxf(v0, v1); __syncthreads();
    for (int off = 128; off > 0; off >>= 1) {
        if (tid < off) red[tid] = fmaxf(red[tid], red[tid + off]);
        __syncthreads();
    }
    float mx = red[0]; __syncthreads();
    float e0 = (tid       < NSEQ) ? expf(v0 - mx) : 0.f;
    float e1 = (tid + 256 < NSEQ) ? expf(v1 - mx) : 0.f;
    red[tid] = e0 + e1; __syncthreads();
    for (int off = 128; off > 0; off >>= 1) {
        if (tid < off) red[tid] += red[tid + off];
        __syncthreads();
    }
    float inv = 1.f / red[0];
    if (tid       < NSEQ) attn[base + tid]       = e0 * inv;
    if (tid + 256 < NSEQ) attn[base + tid + 256] = e1 * inv;
}

// ---------------- o = attn @ v  (K = 273, zero-padded per 32); bf16 out ----------
// K-dim padding feeds every output -> must be zero-filled (no clamping here).
__global__ void __launch_bounds__(256) k_av(const float* __restrict__ attn,
                                            const ushort_t* __restrict__ qkv,
                                            ushort_t* __restrict__ o) {
    __shared__ __attribute__((aligned(16))) unsigned int AsU[64 * 20];  // attn rows
    __shared__ __attribute__((aligned(16))) unsigned int VsT[64 * 20];  // v d-major
    int tid = threadIdx.x, lane = tid & 31, wave = tid >> 5;
    int waveM = wave >> 1, waveN = wave & 1;
    int tI = blockIdx.x * 64;
    int bh = blockIdx.y, b = bh / HEADS_, hd = bh % HEADS_;
    const float* ab = attn + (size_t)bh * NSEQ * NSEQ;
    const ushort_t* vb = qkv + (size_t)b * NSEQ * (3 * DIM_) + 2 * DIM_ + hd * DH_;

    int ar = tid >> 2, aq = tid & 3;               // attn stage: row ar, cols aq*8..+8
    int agi = tI + ar;
    int vd = tid & 63, vq = tid >> 6;              // v stage: col vd, K vq*8..+8

    v8f acc[2] = {};
    for (int k0 = 0; k0 < NSEQ; k0 += 32) {
        {   // attn tile 64x32 f32 -> bf16 (rows not 16B aligned -> scalar loads)
            unsigned int u[4];
            #pragma unroll
            for (int kk = 0; kk < 4; ++kk) {
                int c0 = k0 + aq * 8 + 2 * kk;
                float f0 = (agi < NSEQ && c0     < NSEQ) ? ab[(size_t)agi * NSEQ + c0]     : 0.f;
                float f1 = (agi < NSEQ && c0 + 1 < NSEQ) ? ab[(size_t)agi * NSEQ + c0 + 1] : 0.f;
                u[kk] = pk2(f0, f1);
            }
            *(uint4*)&AsU[ar * 20 + aq * 4] = make_uint4(u[0], u[1], u[2], u[3]);
        }
        {   // v tile 32x64 bf16 -> d-major (u16 gathers, coalesced across vd)
            unsigned int u[4];
            #pragma unroll
            for (int kk = 0; kk < 4; ++kk) {
                int g0 = k0 + vq * 8 + 2 * kk;
                unsigned int v0 = (g0     < NSEQ) ? (unsigned int)vb[(size_t)g0 * (3 * DIM_) + vd]       : 0u;
                unsigned int v1 = (g0 + 1 < NSEQ) ? (unsigned int)vb[(size_t)(g0 + 1) * (3 * DIM_) + vd] : 0u;
                u[kk] = v0 | (v1 << 16);
            }
            *(uint4*)&VsT[vd * 20 + vq * 4] = make_uint4(u[0], u[1], u[2], u[3]);
        }
        __syncthreads();

        Frag16 af, bfv;
        int ku = (lane >> 4) * 4, kbo = (lane >> 4) * 4;
        int row = waveM * 16 + (lane & 15);
        af.q[0] = *(const uint4*)&AsU[row * 20 + ku];
        af.q[1] = *(const uint4*)&AsU[row * 20 + 8 + ku];
        #pragma unroll
        for (int tn = 0; tn < 2; ++tn) {
            int col = waveN * 32 + tn * 16 + (lane & 15);
            bfv.q[0] = *(const uint4*)&VsT[col * 20 + kbo];
            bfv.q[1] = *(const uint4*)&VsT[col * 20 + kbo + 8];
            acc[tn] = __builtin_amdgcn_wmma_f32_16x16x32_bf16(
                false, af.v, false, bfv.v, (short)0, acc[tn], false, false);
        }
        __syncthreads();
    }
    #pragma unroll
    for (int tn = 0; tn < 2; ++tn) {
        int d = waveN * 32 + tn * 16 + (lane & 15);
        #pragma unroll
        for (int r = 0; r < 8; ++r) {
            int gi = tI + waveM * 16 + r + ((lane >> 4) << 3);
            if (gi < NSEQ)
                o[((size_t)(b * NSEQ + gi)) * DIM_ + hd * DH_ + d] = (ushort_t)f2bf(acc[tn][r]);
        }
    }
}

// ---------------- host driver ----------------
extern "C" void kernel_launch(void* const* d_in, const int* in_sizes, int n_in,
                              void* d_out, int out_size, void* d_ws, size_t ws_size,
                              hipStream_t stream) {
    (void)in_sizes; (void)n_in; (void)out_size; (void)ws_size;
    const float* in_x = (const float*)d_in[0];
    const float* pos  = (const float*)d_in[1];
    const float* ln1w = (const float*)d_in[2];
    const float* ln1b = (const float*)d_in[3];
    const float* wqkv = (const float*)d_in[4];
    const float* wout = (const float*)d_in[5];
    const float* bout = (const float*)d_in[6];
    const float* ln2w = (const float*)d_in[7];
    const float* ln2b = (const float*)d_in[8];
    const float* w1   = (const float*)d_in[9];
    const float* b1   = (const float*)d_in[10];
    const float* w2   = (const float*)d_in[11];
    const float* b2   = (const float*)d_in[12];
    float* out = (float*)d_out;

    float*    x    = (float*)d_ws;                               // NROWS*768 f32
    ushort_t* h    = (ushort_t*)(x + (size_t)NROWS * DIM_);      // NROWS*768 bf16
    ushort_t* qkvb = h    + (size_t)NROWS * DIM_;                // NROWS*2304 bf16
    ushort_t* obuf = qkvb + (size_t)NROWS * 3 * DIM_;            // NROWS*768 bf16
    ushort_t* g    = obuf + (size_t)NROWS * DIM_;                // NROWS*3072 bf16
    ushort_t* wt   = g    + (size_t)NROWS * MLP_;                // 768*3072 bf16 (max W)

    float* resOut  = out;
    float* attnOut = out + (size_t)DEPTH_ * NROWS * DIM_;

    hipMemcpyAsync(x, in_x, (size_t)NROWS * DIM_ * sizeof(float),
                   hipMemcpyDeviceToDevice, stream);

    for (int l = 0; l < DEPTH_; ++l) {
        if (l > 0)
            k_addpos<<<(B_ * (NSEQ - J_) * DIM_) / 256, 256, 0, stream>>>(x, pos);

        k_layernorm<<<NROWS, 256, 0, stream>>>(x, ln1w + l * DIM_, ln1b + l * DIM_, h);

        // qkv = h @ w_qkv[l]   [4368x768]@[768x2304] -> bf16
        k_wtrans<<<dim3((3 * DIM_) / 32, DIM_ / 32), 256, 0, stream>>>(
            wqkv + (size_t)l * DIM_ * 3 * DIM_, wt, DIM_, 3 * DIM_);
        k_gemm<<<dim3((3 * DIM_) / 64, (NROWS + 127) / 128), 256, 0, stream>>>(
            h, wt, nullptr, nullptr, nullptr, qkvb, NROWS, DIM_, 3 * DIM_, 0);

        float* attnL = attnOut + (size_t)l * B_ * HEADS_ * NSEQ * NSEQ;
        k_scores<<<dim3(5, 5, B_ * HEADS_), 256, 0, stream>>>(qkvb, attnL);
        k_softmax<<<B_ * HEADS_ * NSEQ, 256, 0, stream>>>(attnL);
        k_av<<<dim3(5, B_ * HEADS_), 256, 0, stream>>>(attnL, qkvb, obuf);

        // x = obuf @ w_out + b_out + x
        k_wtrans<<<dim3(DIM_ / 32, DIM_ / 32), 256, 0, stream>>>(
            wout + (size_t)l * DIM_ * DIM_, wt, DIM_, DIM_);
        k_gemm<<<dim3(DIM_ / 64, (NROWS + 127) / 128), 256, 0, stream>>>(
            obuf, wt, bout + (size_t)l * DIM_, x, x, nullptr, NROWS, DIM_, DIM_, 0);

        k_layernorm<<<NROWS, 256, 0, stream>>>(x, ln2w + l * DIM_, ln2b + l * DIM_, h);

        // g = gelu(h @ w1 + b1) -> bf16
        k_wtrans<<<dim3(MLP_ / 32, DIM_ / 32), 256, 0, stream>>>(
            w1 + (size_t)l * DIM_ * MLP_, wt, DIM_, MLP_);
        k_gemm<<<dim3(MLP_ / 64, (NROWS + 127) / 128), 256, 0, stream>>>(
            h, wt, b1 + (size_t)l * MLP_, nullptr, nullptr, g, NROWS, DIM_, MLP_, 1);

        // x = g @ w2 + b2 + x
        k_wtrans<<<dim3(DIM_ / 32, MLP_ / 32), 256, 0, stream>>>(
            w2 + (size_t)l * MLP_ * DIM_, wt, MLP_, DIM_);
        k_gemm<<<dim3(DIM_ / 64, (NROWS + 127) / 128), 256, 0, stream>>>(
            g, wt, b2 + (size_t)l * DIM_, x, x, nullptr, NROWS, MLP_, DIM_, 0);

        hipMemcpyAsync(resOut + (size_t)l * NROWS * DIM_, x,
                       (size_t)NROWS * DIM_ * sizeof(float),
                       hipMemcpyDeviceToDevice, stream);
    }
}
